// EquivariantThreeHopGINE_36713380446554
// MI455X (gfx1250) — matrix operand order repaired and checked by reading, not compiled, not run on hardware
//
#include <hip/hip_runtime.h>
#include <hip/hip_bf16.h>

#define N_NODES 100000
#define N_EDGES 300000
#define HID 256

typedef __attribute__((ext_vector_type(16))) __bf16 v16bf;
typedef __attribute__((ext_vector_type(8)))  float  v8f;

__device__ __forceinline__ unsigned short f2bf(float f) {
    unsigned u = __builtin_bit_cast(unsigned, f);
    unsigned r = u + 0x7FFFu + ((u >> 16) & 1u);   // round-to-nearest-even
    return (unsigned short)(r >> 16);
}

// ---------------------------------------------------------------------------
// f32 -> bf16 weight conversion
// ---------------------------------------------------------------------------
__global__ void cvt_bf16_kernel(const float* __restrict__ src,
                                unsigned short* __restrict__ dst, int n) {
    int i = blockIdx.x * blockDim.x + threadIdx.x;
    if (i < n) dst[i] = f2bf(src[i]);
}

// ---------------------------------------------------------------------------
// Per-edge-type lift table: elift[t][o] = b[o] + sum_k bond_emb[t][k]*W[o][k]
// grid = 5 (types), block = 256 (outputs)
// ---------------------------------------------------------------------------
__global__ void elift_kernel(const float* __restrict__ bond_emb,   // [5,32]
                             const float* __restrict__ W,          // [256,32]
                             const float* __restrict__ b,          // [256]
                             float* __restrict__ out) {            // [5,256]
    int t = blockIdx.x, o = threadIdx.x;
    float acc = b[o];
    #pragma unroll
    for (int k = 0; k < 32; ++k) acc += bond_emb[t * 32 + k] * W[o * 32 + k];
    out[t * 256 + o] = acc;
}

// ---------------------------------------------------------------------------
// Atom embedding: one thread per node -> feat [N,64]
// ---------------------------------------------------------------------------
__global__ void atom_embed_kernel(
    const float* __restrict__ x,                                    // [N,79]
    const float* __restrict__ el,  const float* __restrict__ deg,
    const float* __restrict__ ring, const float* __restrict__ chg,
    const float* __restrict__ aro, const float* __restrict__ hyb,
    const float* __restrict__ hyd, const float* __restrict__ func,  // [18,2,2]
    const float* __restrict__ hdon, const float* __restrict__ hacc,
    const float* __restrict__ ringsz, const float* __restrict__ aronum,
    const float* __restrict__ fused, const float* __restrict__ het,
    const float* __restrict__ frW, const float* __restrict__ frB,   // [4,36],[4]
    const float* __restrict__ beW, const float* __restrict__ beB,   // [16,48],[16]
    const float* __restrict__ dW,  const float* __restrict__ dB,    // [48,52],[48]
    float* __restrict__ feat, int n) {
    int i = blockIdx.x * blockDim.x + threadIdx.x;
    if (i >= n) return;
    const float* xi = x + (long)i * 79;

    float d52[52];
    const int LUT[17] = {0,0,0,0,0,0,1,2,3,0,0,0,0,0,4,5,6};
    int z = (int)xi[0]; if (z < 0 || z > 16) z = 0;
    int i0 = LUT[z];
    int c1 = (int)xi[1]; c1 = c1 < 0 ? 0 : (c1 > 6 ? 6 : c1);
    int c2 = (int)xi[5] + 1; c2 = c2 < 0 ? 0 : (c2 > 1 ? 1 : c2);
    int c3 = (int)xi[2]; c3 = c3 < 0 ? 0 : (c3 > 7 ? 7 : c3);
    int c4 = (int)xi[4]; c4 = c4 < 0 ? 0 : (c4 > 1 ? 1 : c4);
    int c5 = (int)xi[3]; c5 = c5 < 0 ? 0 : (c5 > 5 ? 5 : c5);
    int c6 = (int)xi[6]; c6 = c6 < 0 ? 0 : (c6 > 4 ? 4 : c6);
    #pragma unroll
    for (int c = 0; c < 4; ++c) {
        d52[0  + c] = el [i0 * 4 + c];
        d52[4  + c] = deg[c1 * 4 + c];
        d52[8  + c] = ring[c2 * 4 + c];
        d52[12 + c] = chg[c3 * 4 + c];
        d52[16 + c] = aro[c4 * 4 + c];
        d52[20 + c] = hyb[c5 * 4 + c];
        d52[24 + c] = hyd[c6 * 4 + c];
    }
    // 18 binary flag tables -> 36-vector -> 4 via func_reduce
    float fl[36];
    #pragma unroll
    for (int j = 0; j < 18; ++j) {
        int f = (int)xi[7 + j]; f = f < 0 ? 0 : (f > 1 ? 1 : f);
        fl[2 * j]     = func[j * 4 + f * 2];
        fl[2 * j + 1] = func[j * 4 + f * 2 + 1];
    }
    for (int o = 0; o < 4; ++o) {
        float acc = frB[o];
        #pragma unroll
        for (int k = 0; k < 36; ++k) acc += fl[k] * frW[o * 36 + k];
        d52[28 + o] = acc;
    }
    int c25 = (int)xi[25]; c25 = c25 < 0 ? 0 : (c25 > 1 ? 1 : c25);
    int c26 = (int)xi[26]; c26 = c26 < 0 ? 0 : (c26 > 1 ? 1 : c26);
    d52[32] = hdon[c25 * 2]; d52[33] = hdon[c25 * 2 + 1];
    d52[34] = hacc[c26 * 2]; d52[35] = hacc[c26 * 2 + 1];
    const int RV[7] = {0,3,4,5,6,7,8};
    int raw27 = (int)xi[27], m27 = 6;
    #pragma unroll
    for (int t = 6; t >= 0; --t) if (raw27 == RV[t]) m27 = t;  // first match wins
    int c28 = (int)xi[28]; c28 = c28 < 0 ? 0 : (c28 > 4 ? 4 : c28);
    int c29 = (int)xi[29]; c29 = c29 < 0 ? 0 : (c29 > 7 ? 7 : c29);
    int c30 = (int)xi[30]; c30 = c30 < 0 ? 0 : (c30 > 26 ? 26 : c30);
    #pragma unroll
    for (int c = 0; c < 4; ++c) {
        d52[36 + c] = ringsz[m27 * 4 + c];
        d52[40 + c] = aronum[c28 * 4 + c];
        d52[44 + c] = fused [c29 * 4 + c];
        d52[48 + c] = het   [c30 * 4 + c];
    }
    float* fo = feat + (long)i * 64;
    for (int o = 0; o < 48; ++o) {
        float acc = dB[o];
        #pragma unroll
        for (int k = 0; k < 52; ++k) acc += d52[k] * dW[o * 52 + k];
        fo[o] = acc;
    }
    for (int o = 0; o < 16; ++o) {
        float acc = beB[o];
        #pragma unroll
        for (int k = 0; k < 48; ++k) acc += xi[31 + k] * beW[o * 48 + k];
        fo[48 + o] = acc;
    }
}

// ---------------------------------------------------------------------------
// WMMA GEMM: Cout[M,256] = act(A_f32[M,K] @ Wbf16[256,ldb]^T [+ bias] [+ Cin])
// grid.x = M/16 row tiles, block = 256 threads (8 waves), wave w -> cols
// [w*32, w*32+32). K-loop step 32, two v_wmma_f32_16x16x32_bf16 per step,
// unrolled 2x (K/32 is always even here). Flags are compile-time so the
// epilogue is branch-free straight-line code.
// ---------------------------------------------------------------------------
template <bool BIAS, bool CIN, bool RELU>
__global__ void __launch_bounds__(256) gemm_wmma_kernel(
    const float* __restrict__ A, int lda,
    const unsigned short* __restrict__ Wb, int ldb,
    const float* __restrict__ bias,
    const float* __restrict__ Cin,
    float* __restrict__ Cout,
    int K) {
    const int lane = threadIdx.x & 31;
    const int wv   = threadIdx.x >> 5;     // column tile pair 0..7
    const int rt   = blockIdx.x;           // row tile
    const int half = lane >> 4;
    const int ml   = lane & 15;

    const float* Arow = A + (long)(rt * 16 + ml) * lda;
    const int ncol0 = wv * 32 + ml;
    const unsigned short* Bp0 = Wb + (long)ncol0 * ldb + half * 16;
    const unsigned short* Bp1 = Wb + (long)(ncol0 + 16) * ldb + half * 16;

    v8f acc0 = {}; v8f acc1 = {};

    #pragma unroll 2
    for (int k0 = 0; k0 < K; k0 += 32) {
        // A fragment: lane half selects K offsets 0..15 + 16*half
        union { v16bf v; __bf16 h[16]; } af;
        const float4* ap0 = (const float4*)(Arow + k0 + half * 16);
        float4 a0 = ap0[0], a1 = ap0[1], a2 = ap0[2], a3 = ap0[3];
        float av[16] = {a0.x,a0.y,a0.z,a0.w, a1.x,a1.y,a1.z,a1.w,
                        a2.x,a2.y,a2.z,a2.w, a3.x,a3.y,a3.z,a3.w};
        #pragma unroll
        for (int e = 0; e < 16; ++e) af.h[e] = (__bf16)av[e];  // native cvt

        // B fragments: 16 contiguous bf16 (one row of W = one col of W^T)
        union { v16bf v; uint4 q[2]; } bf0, bf1;
        const uint4* bq0 = (const uint4*)(Bp0 + k0);
        bf0.q[0] = bq0[0]; bf0.q[1] = bq0[1];
        const uint4* bq1 = (const uint4*)(Bp1 + k0);
        bf1.q[0] = bq1[0]; bf1.q[1] = bq1[1];

        acc0 = __builtin_amdgcn_wmma_f32_16x16x32_bf16(
                   false, af.v, false, bf0.v, (short)0, acc0, false, false);
        acc1 = __builtin_amdgcn_wmma_f32_16x16x32_bf16(
                   false, af.v, false, bf1.v, (short)0, acc1, false, false);
    }

    // epilogue: vgpr r holds row rt*16 + r + 8*half, col = ncol0 (+16)
    #pragma unroll
    for (int t = 0; t < 2; ++t) {
        int col = ncol0 + t * 16;
        float bv = BIAS ? bias[col] : 0.0f;
        v8f acc = t ? acc1 : acc0;
        #pragma unroll
        for (int r = 0; r < 8; ++r) {
            long orow = (long)(rt * 16 + r + 8 * half);
            float v = acc[r] + bv;
            if (CIN)  v += Cin[orow * 256 + col];
            if (RELU) v = fmaxf(v, 0.0f);
            Cout[orow * 256 + col] = v;
        }
    }
}

// ---------------------------------------------------------------------------
// LayerNorm (wave per row, 8 floats/lane, shfl_xor reductions)
// mode 0: Y = LN(X)*g+b              (ln_in -> h0)
// mode 1: H = r*(LN(X)*g+b) + (1-r)*H (gated residual, in-place on H)
// ---------------------------------------------------------------------------
__global__ void ln_kernel(const float* __restrict__ X,
                          const float* __restrict__ g,
                          const float* __restrict__ b,
                          const float* __restrict__ res,   // scalar or null
                          float* __restrict__ H, int nrows, int mode) {
    int w = blockIdx.x * (blockDim.x >> 5) + (threadIdx.x >> 5);
    if (w >= nrows) return;
    int lane = threadIdx.x & 31;
    const float4* p = (const float4*)(X + (long)w * 256 + lane * 8);
    float4 a = p[0], c = p[1];
    float v[8] = {a.x,a.y,a.z,a.w, c.x,c.y,c.z,c.w};
    float s = 0.f;
    #pragma unroll
    for (int j = 0; j < 8; ++j) s += v[j];
    #pragma unroll
    for (int m = 1; m < 32; m <<= 1) s += __shfl_xor(s, m, 32);
    float mean = s * (1.0f / 256.0f);
    float q = 0.f;
    #pragma unroll
    for (int j = 0; j < 8; ++j) { float d = v[j] - mean; q += d * d; }
    #pragma unroll
    for (int m = 1; m < 32; m <<= 1) q += __shfl_xor(q, m, 32);
    float rstd = rsqrtf(q * (1.0f / 256.0f) + 1e-5f);

    float* hp = H + (long)w * 256 + lane * 8;
    float r = (mode == 1) ? *res : 1.0f;
    float out[8];
    #pragma unroll
    for (int j = 0; j < 8; ++j) {
        float y = (v[j] - mean) * rstd * g[lane * 8 + j] + b[lane * 8 + j];
        out[j] = (mode == 1) ? (r * y + (1.0f - r) * hp[j]) : y;
    }
    float4* op = (float4*)hp;
    op[0] = make_float4(out[0], out[1], out[2], out[3]);
    op[1] = make_float4(out[4], out[5], out[6], out[7]);
}

// ---------------------------------------------------------------------------
// Message passing: agg[dst] += relu(h[src] + elift[type])  (agg pre-set to h)
// one wave per edge, 8 floats/lane, hardware f32 global atomics
// ---------------------------------------------------------------------------
__global__ void msg_kernel(const float* __restrict__ h,
                           const float* __restrict__ elift,   // [5,256]
                           const int* __restrict__ src,
                           const int* __restrict__ dst,
                           const int* __restrict__ etype,
                           float* __restrict__ agg, int nE) {
    int e = blockIdx.x * (blockDim.x >> 5) + (threadIdx.x >> 5);
    if (e >= nE) return;
    int lane = threadIdx.x & 31;
    int s = src[e], d = dst[e], t = etype[e];
    t = t < 0 ? 0 : (t > 4 ? 4 : t);
    int c = lane * 8;
    const float4* hp = (const float4*)(h + (long)s * 256 + c);
    const float4* ep = (const float4*)(elift + t * 256 + c);
    float4 h0 = hp[0], h1 = hp[1], e0 = ep[0], e1 = ep[1];
    float m[8] = {h0.x + e0.x, h0.y + e0.y, h0.z + e0.z, h0.w + e0.w,
                  h1.x + e1.x, h1.y + e1.y, h1.z + e1.z, h1.w + e1.w};
    float* ap = agg + (long)d * 256 + c;
    #pragma unroll
    for (int j = 0; j < 8; ++j) unsafeAtomicAdd(ap + j, fmaxf(m[j], 0.0f));
}

// ---------------------------------------------------------------------------
extern "C" void kernel_launch(void* const* d_in, const int* in_sizes, int n_in,
                              void* d_out, int out_size, void* d_ws, size_t ws_size,
                              hipStream_t stream) {
    (void)in_sizes; (void)n_in; (void)out_size; (void)ws_size;
    const float* atom  = (const float*)d_in[0];
    const int*   eidx  = (const int*)d_in[1];
    const int*   etype = (const int*)d_in[2];
    const int* src = eidx;
    const int* dst = eidx + N_EDGES;

    // params (setup_inputs insertion order, tuples flattened (W,b))
    int pi = 3;
    const float* P[54];
    for (int i = 0; i < 54; ++i) P[i] = (const float*)d_in[pi + i];
    const float *el = P[0], *deg = P[1], *ring = P[2], *chg = P[3], *aro = P[4],
                *hyb = P[5], *hyd = P[6], *func = P[7], *hdon = P[8], *hacc = P[9],
                *ringsz = P[10], *aronum = P[11], *fused = P[12], *het = P[13],
                *frW = P[14], *frB = P[15], *beW = P[16], *beB = P[17],
                *dW = P[18], *dB = P[19], *W0 = P[20], *b0 = P[21],
                *lnInG = P[22], *lnInB = P[23], *bondEmb = P[24];
    const float *gineLinW[3], *gineLinB[3], *W1[3], *b1[3], *W2[3], *b2[3],
                *lnG[3], *lnB[3], *resP[3];
    for (int l = 0; l < 3; ++l) {
        int base = 25 + l * 9;
        gineLinW[l] = P[base + 0]; gineLinB[l] = P[base + 1];
        W1[l] = P[base + 2]; b1[l] = P[base + 3];
        W2[l] = P[base + 4]; b2[l] = P[base + 5];
        lnG[l] = P[base + 6]; lnB[l] = P[base + 7]; resP[l] = P[base + 8];
    }
    const float* jkW = P[52];
    const float* jkB = P[53];

    // workspace layout
    float* feat = (float*)d_ws;                       // N*64
    float* h    = feat + (size_t)N_NODES * 64;        // N*256
    float* agg  = h    + (size_t)N_NODES * 256;       // N*256
    float* tA   = agg  + (size_t)N_NODES * 256;       // N*256
    float* tB   = tA   + (size_t)N_NODES * 256;       // N*256
    unsigned short* W0b = (unsigned short*)(tB + (size_t)N_NODES * 256); // 256*64
    unsigned short* W1b = W0b + 256 * 64;             // 3 * 256*256
    unsigned short* W2b = W1b + 3 * 256 * 256;        // 3 * 256*256
    unsigned short* jkb = W2b + 3 * 256 * 256;        // 256*1024
    float* elift = (float*)(jkb + 256 * 1024);        // 3 * 5*256
    float* outF  = (float*)d_out;

    // 1. weight conversion to bf16
    cvt_bf16_kernel<<<(256 * 64 + 255) / 256, 256, 0, stream>>>(W0, W0b, 256 * 64);
    for (int l = 0; l < 3; ++l) {
        cvt_bf16_kernel<<<(65536 + 255) / 256, 256, 0, stream>>>(W1[l], W1b + l * 65536, 65536);
        cvt_bf16_kernel<<<(65536 + 255) / 256, 256, 0, stream>>>(W2[l], W2b + l * 65536, 65536);
    }
    cvt_bf16_kernel<<<(256 * 1024 + 255) / 256, 256, 0, stream>>>(jkW, jkb, 256 * 1024);

    // 2. edge-type lift tables (5 types x 256 per layer)
    for (int l = 0; l < 3; ++l)
        elift_kernel<<<5, 256, 0, stream>>>(bondEmb, gineLinW[l], gineLinB[l],
                                            elift + l * 5 * 256);

    // 3. atom embedding -> feat [N,64]
    atom_embed_kernel<<<(N_NODES + 255) / 256, 256, 0, stream>>>(
        atom, el, deg, ring, chg, aro, hyb, hyd, func, hdon, hacc,
        ringsz, aronum, fused, het, frW, frB, beW, beB, dW, dB, feat, N_NODES);

    const int GB = N_NODES / 16;       // 6250 row tiles, exact
    const int LB = N_NODES / 8;        // 12500 LN blocks, exact
    const int MB = N_EDGES / 8;        // 37500 msg blocks, exact

    // 4. h0 = LN(feat @ W0^T + b0)
    gemm_wmma_kernel<true, false, false><<<GB, 256, 0, stream>>>(
        feat, 64, W0b, 64, b0, nullptr, tB, 64);
    ln_kernel<<<LB, 256, 0, stream>>>(tB, lnInG, lnInB, nullptr, h, N_NODES, 0);
    // jk partial 0: out = jkB + h0 @ jkW[:, 0:256]^T  (initializes d_out)
    gemm_wmma_kernel<true, false, false><<<GB, 256, 0, stream>>>(
        h, 256, jkb, 1024, jkB, nullptr, outF, 256);

    // 5. three GINE layers
    for (int l = 0; l < 3; ++l) {
        hipMemcpyAsync(agg, h, (size_t)N_NODES * 256 * sizeof(float),
                       hipMemcpyDeviceToDevice, stream);                    // agg = x
        msg_kernel<<<MB, 256, 0, stream>>>(h, elift + l * 5 * 256, src, dst,
                                           etype, agg, N_EDGES);            // += msgs
        gemm_wmma_kernel<true, false, true><<<GB, 256, 0, stream>>>(
            agg, 256, W1b + l * 65536, 256, b1[l], nullptr, tA, 256);
        gemm_wmma_kernel<true, false, true><<<GB, 256, 0, stream>>>(
            tA, 256, W2b + l * 65536, 256, b2[l], nullptr, tB, 256);
        ln_kernel<<<LB, 256, 0, stream>>>(tB, lnG[l], lnB[l], resP[l], h, N_NODES, 1);
        // jk partial (l+1): out += h @ jkW[:, (l+1)*256 : (l+2)*256]^T
        gemm_wmma_kernel<false, true, false><<<GB, 256, 0, stream>>>(
            h, 256, jkb + (l + 1) * 256, 1024, nullptr, outF, outF, 256);
    }
}
